// Encoder_59356448031597
// MI455X (gfx1250) — compile-verified, matrix operand order checked by reading, not run on hardware
//
#include <hip/hip_runtime.h>
#include <stddef.h>

// ---------------- problem constants ----------------
#define BB   256          // batch
#define TT   361          // input time
#define DD   128          // input dim
#define HH   512          // hidden
#define SS   40           // scales
#define ML   88           // max_len: ((361-2)/2 - 2)/2 = 88
#define NH   1280         // 32 * SS
#define G4   2048         // 4 * HH
#define MROWS (BB * ML)   // 22528 rows of the big GEMM

// ---------------- WMMA plumbing ----------------
typedef __attribute__((ext_vector_type(16))) __bf16 v16bf;
typedef __attribute__((ext_vector_type(8)))  float  v8f;

struct alignas(16) US8 { unsigned short x[8]; };

__device__ __forceinline__ unsigned short f32_to_bf16_rne(float f) {
    unsigned int u = __builtin_bit_cast(unsigned int, f);
    unsigned int lsb = (u >> 16) & 1u;
    u += 0x7fffu + lsb;
    return (unsigned short)(u >> 16);
}

// Fragment loader: p = row_base + k0 + (lane>>4)*8.
// Elements 0-7 <- p[0..7], elements 8-15 <- p[16..23]  (two b128 loads).
__device__ __forceinline__ v16bf load_frag(const unsigned short* p) {
    union { US8 h[2]; v16bf v; } u;
    u.h[0] = *(const US8*)(p);
    u.h[1] = *(const US8*)(p + 16);
    return u.v;
}

__device__ __forceinline__ v8f wmma_bf16(v16bf a, v16bf b, v8f c) {
    // (neg_a, A, neg_b, B, c_mod, C, reuse_a, reuse_b)
    return __builtin_amdgcn_wmma_f32_16x16x32_bf16(false, a, false, b, (short)0, c, false, false);
}

__device__ __forceinline__ float sigmoidf_(float x) { return 1.0f / (1.0f + __expf(-x)); }

// ---------------- utility kernels ----------------
__global__ void zero_u32_k(unsigned int* p, long long n) {
    long long i = (long long)blockIdx.x * blockDim.x + threadIdx.x;
    long long st = (long long)gridDim.x * blockDim.x;
    for (; i < n; i += st) p[i] = 0u;
}

__global__ void cvt_bf16_k(const float* s, unsigned short* d, long long n) {
    long long i = (long long)blockIdx.x * blockDim.x + threadIdx.x;
    long long st = (long long)gridDim.x * blockDim.x;
    for (; i < n; i += st) d[i] = f32_to_bf16_rne(s[i]);
}

__global__ void bias_k(const float* a, const float* b, float* o) {
    int i = blockIdx.x * blockDim.x + threadIdx.x;
    if (i < G4) o[i] = a[i] + b[i];
}

// ---------------- conv pyramid: one block per (scale, batch) ----------------
__global__ __launch_bounds__(256)
void conv_pyramid_k(const float* __restrict__ X,
                    const float* __restrict__ c1w, const float* __restrict__ c1b,
                    const float* __restrict__ c2w, const float* __restrict__ c2b,
                    const float* __restrict__ c3w, const float* __restrict__ c3b,
                    unsigned short* __restrict__ xf) {
    __shared__ alignas(16) float w1s[16 * 128];   // 8 KB
    __shared__ alignas(16) float w2s[32 * 48];    // 6 KB
    __shared__ alignas(16) float w3s[32 * 96];    // 12 KB
    __shared__ alignas(16) float y1[361 * 16];    // 23 KB
    __shared__ alignas(16) float y2[180 * 32];    // 23 KB

    const int sIdx = blockIdx.x;          // 0..39
    const int s    = sIdx + 1;
    const int b    = blockIdx.y;
    const int tid  = threadIdx.x;

    const int L1  = (TT + s - 1) / s;
    const int L2p = (L1 - 2) >> 1;
    const int L3p = (L2p - 2) >> 1;

    // stage weights into LDS
    for (int i = tid; i < 16 * 128; i += 256) w1s[i] = c1w[sIdx * 16 * 128 + i];
    for (int i = tid; i < 32 * 48;  i += 256) w2s[i] = c2w[sIdx * 32 * 48 + i];
    for (int i = tid; i < 32 * 96;  i += 256) w3s[i] = c3w[sIdx * 32 * 96 + i];
    __syncthreads();

    // conv1: pointwise 128 -> 16 on strided input
    const float* xb = X + (size_t)b * TT * DD;
    for (int idx = tid; idx < L1 * 16; idx += 256) {
        const int l = idx >> 4, o = idx & 15;
        const float4* xr = (const float4*)(xb + (size_t)(l * s) * DD);
        const float4* wr = (const float4*)(w1s + o * 128);
        float acc = 0.f;
#pragma unroll 8
        for (int q = 0; q < 32; ++q) {
            float4 xv = xr[q], wv = wr[q];
            acc += xv.x * wv.x + xv.y * wv.y + xv.z * wv.z + xv.w * wv.w;
        }
        y1[l * 16 + o] = acc + c1b[sIdx * 16 + o];
    }
    __syncthreads();

    // conv2 (16->32, k=3, VALID) + maxpool2
    for (int idx = tid; idx < L2p * 32; idx += 256) {
        const int j2 = idx >> 5, o2 = idx & 31;
        const float bv = c2b[sIdx * 32 + o2];
        float m = -3.402823e38f;
#pragma unroll
        for (int p = 0; p < 2; ++p) {
            const int pos = 2 * j2 + p;
            float acc = bv;
#pragma unroll
            for (int k = 0; k < 3; ++k) {
                const float* yr = y1 + (pos + k) * 16;
                const float* wr = w2s + o2 * 48 + k;
#pragma unroll
                for (int o = 0; o < 16; ++o) acc += yr[o] * wr[o * 3];
            }
            m = fmaxf(m, acc);
        }
        y2[j2 * 32 + o2] = m;
    }
    __syncthreads();

    // conv3 (32->32, k=3, VALID) + maxpool2, write bf16 with left padding
    for (int idx = tid; idx < L3p * 32; idx += 256) {
        const int j3 = idx >> 5, o3 = idx & 31;
        const float bv = c3b[sIdx * 32 + o3];
        float m = -3.402823e38f;
#pragma unroll
        for (int p = 0; p < 2; ++p) {
            const int pos = 2 * j3 + p;
            float acc = bv;
#pragma unroll
            for (int k = 0; k < 3; ++k) {
                const float* yr = y2 + (pos + k) * 32;
                const float* wr = w3s + o3 * 96 + k;
#pragma unroll
                for (int o = 0; o < 32; ++o) acc += yr[o] * wr[o * 3];
            }
            m = fmaxf(m, acc);
        }
        const int trow = ML - L3p + j3;
        xf[((size_t)b * ML + trow) * NH + 32 * sIdx + o3] = f32_to_bf16_rne(m);
    }
}

// ---------------- big GEMM: Xin = Xf @ W_ih^T + bias  (22528 x 2048 x 1280) --------
// block = 256 thr (8 waves: 4M x 2N); wave tile = 32M x 64N.
// Per k-step: 4 A b128 + 8 B b128 loads feed 8 WMMAs (1.5 loads/wmma).
__global__ __launch_bounds__(256)
void gemm_in_k(const unsigned short* __restrict__ A,   // [MROWS, NH] bf16
               const unsigned short* __restrict__ W,   // [G4, NH]    bf16
               const float* __restrict__ bias,         // [G4]
               float* __restrict__ Xin) {              // [MROWS, G4]
    const int lane = threadIdx.x & 31;
    const int w    = threadIdx.x >> 5;
    const int m0   = blockIdx.x * 128 + (w & 3) * 32;
    const int n0   = blockIdx.y * 128 + (w >> 2) * 64;

    const v8f vz = {0.f,0.f,0.f,0.f,0.f,0.f,0.f,0.f};
    v8f acc[2][4] = {{vz, vz, vz, vz}, {vz, vz, vz, vz}};

    const int koff = (lane >> 4) * 8;
    const unsigned short* arow0 = A + (size_t)(m0 + (lane & 15)) * NH + koff;
    const unsigned short* arow1 = arow0 + (size_t)16 * NH;
    const unsigned short* brow[4];
#pragma unroll
    for (int j = 0; j < 4; ++j)
        brow[j] = W + (size_t)(n0 + j * 16 + (lane & 15)) * NH + koff;

    for (int k0 = 0; k0 < NH; k0 += 32) {
        v16bf a0 = load_frag(arow0 + k0);
        v16bf a1 = load_frag(arow1 + k0);
#pragma unroll
        for (int j = 0; j < 4; ++j) {
            v16bf bfrag = load_frag(brow[j] + k0);
            acc[0][j] = wmma_bf16(a0, bfrag, acc[0][j]);
            acc[1][j] = wmma_bf16(a1, bfrag, acc[1][j]);
        }
    }

#pragma unroll
    for (int i = 0; i < 2; ++i) {
        const int mbase = m0 + i * 16 + 8 * (lane >> 4);
#pragma unroll
        for (int j = 0; j < 4; ++j) {
            const int n = n0 + j * 16 + (lane & 15);
            const float bn = bias[n];
#pragma unroll
            for (int r = 0; r < 8; ++r)
                Xin[(size_t)(mbase + r) * G4 + n] = acc[i][j][r] + bn;
        }
    }
}

// ---------------- fused LSTM step: gates = Xin[t] + h @ W_hh^T, then cell ----------
// grid (4, 32), block 128 (4 waves). Wave: 16 batch x 16 hidden, all 4 gates.
// Xin / c_prev loads issued BEFORE the K loop to overlap with the 64 WMMAs
// (this kernel runs 88x serially -> hide everything possible under the GEMM).
__global__ __launch_bounds__(128)
void lstm_step_k(const float* __restrict__ Xin,          // [MROWS, G4]
                 const unsigned short* __restrict__ Whh, // [G4, HH] bf16
                 const unsigned short* __restrict__ hp,  // [BB, HH] bf16
                 unsigned short* __restrict__ hn,        // [BB, HH] bf16
                 float* __restrict__ cbuf,               // [BB, HH]
                 float* __restrict__ out,                // [BB, ML, HH]
                 int t) {
    const int lane = threadIdx.x & 31;
    const int w    = threadIdx.x >> 5;
    const int m0   = blockIdx.x * 64 + w * 16;   // batch tile
    const int n0   = blockIdx.y * 16;            // hidden tile

    const int nh = n0 + (lane & 15);
    const int mb = m0 + 8 * (lane >> 4);

    // ---- prefetch gate biases from Xin[t] and previous cell state ----
    float xi[8], xfg[8], xg[8], xo[8], cprev[8];
#pragma unroll
    for (int r = 0; r < 8; ++r) {
        const int b = mb + r;
        const size_t xb = ((size_t)b * ML + t) * G4;
        xi[r]    = Xin[xb + 0 * HH + nh];
        xfg[r]   = Xin[xb + 1 * HH + nh];
        xg[r]    = Xin[xb + 2 * HH + nh];
        xo[r]    = Xin[xb + 3 * HH + nh];
        cprev[r] = cbuf[(size_t)b * HH + nh];
    }

    // ---- recurrent GEMM: all four gate tiles share each A fragment ----
    const v8f vz = {0.f,0.f,0.f,0.f,0.f,0.f,0.f,0.f};
    v8f acc[4] = {vz, vz, vz, vz};               // i, f, g, o

    const int koff = (lane >> 4) * 8;
    const unsigned short* arow = hp + (size_t)(m0 + (lane & 15)) * HH + koff;
    const unsigned short* brow[4];
#pragma unroll
    for (int g = 0; g < 4; ++g)
        brow[g] = Whh + (size_t)(g * HH + n0 + (lane & 15)) * HH + koff;

    for (int k0 = 0; k0 < HH; k0 += 32) {
        v16bf a = load_frag(arow + k0);
#pragma unroll
        for (int g = 0; g < 4; ++g)
            acc[g] = wmma_bf16(a, load_frag(brow[g] + k0), acc[g]);
    }

    // ---- cell update (lane-local: all four gate tiles share D indexing) ----
#pragma unroll
    for (int r = 0; r < 8; ++r) {
        const int b = mb + r;
        const float i = sigmoidf_(acc[0][r] + xi[r]);
        const float f = sigmoidf_(acc[1][r] + xfg[r]);
        const float g = tanhf(acc[2][r] + xg[r]);
        const float o = sigmoidf_(acc[3][r] + xo[r]);
        const size_t ci = (size_t)b * HH + nh;
        const float c = f * cprev[r] + i * g;
        cbuf[ci] = c;
        const float h = o * tanhf(c);
        out[((size_t)b * ML + t) * HH + nh] = h;
        hn[ci] = f32_to_bf16_rne(h);
    }
}

// ---------------- host launcher ----------------
extern "C" void kernel_launch(void* const* d_in, const int* in_sizes, int n_in,
                              void* d_out, int out_size, void* d_ws, size_t ws_size,
                              hipStream_t stream) {
    const float* X   = (const float*)d_in[0];
    const float* c1w = (const float*)d_in[1];
    const float* c1b = (const float*)d_in[2];
    const float* c2w = (const float*)d_in[3];
    const float* c2b = (const float*)d_in[4];
    const float* c3w = (const float*)d_in[5];
    const float* c3b = (const float*)d_in[6];
    const float* wih = (const float*)d_in[7];
    const float* whh = (const float*)d_in[8];
    const float* bih = (const float*)d_in[9];
    const float* bhh = (const float*)d_in[10];
    float* out = (float*)d_out;

    // workspace layout (256B aligned)
    char* ws = (char*)d_ws;
    auto align_up = [](size_t x) { return (x + 255) & ~(size_t)255; };
    size_t cur = 0;
    const size_t XF_B   = (size_t)MROWS * NH * sizeof(unsigned short);
    const size_t WIH_B  = (size_t)G4 * NH * sizeof(unsigned short);
    const size_t WHH_B  = (size_t)G4 * HH * sizeof(unsigned short);
    const size_t BIAS_B = (size_t)G4 * sizeof(float);
    const size_t XIN_B  = (size_t)MROWS * G4 * sizeof(float);
    const size_t H_B    = (size_t)BB * HH * sizeof(unsigned short);
    const size_t C_B    = (size_t)BB * HH * sizeof(float);

    size_t off_xf   = cur; cur = align_up(cur + XF_B);
    size_t off_wih  = cur; cur = align_up(cur + WIH_B);
    size_t off_whh  = cur; cur = align_up(cur + WHH_B);
    size_t off_bias = cur; cur = align_up(cur + BIAS_B);
    size_t off_xin  = cur; cur = align_up(cur + XIN_B);
    size_t off_hA   = cur; cur = align_up(cur + H_B);
    size_t off_hB   = cur; cur = align_up(cur + H_B);
    size_t off_c    = cur; cur = align_up(cur + C_B);
    (void)ws_size; (void)n_in; (void)in_sizes; (void)out_size;

    unsigned short* xf_bf  = (unsigned short*)(ws + off_xf);
    unsigned short* wih_bf = (unsigned short*)(ws + off_wih);
    unsigned short* whh_bf = (unsigned short*)(ws + off_whh);
    float*          biasws = (float*)(ws + off_bias);
    float*          xin    = (float*)(ws + off_xin);
    unsigned short* hA     = (unsigned short*)(ws + off_hA);
    unsigned short* hB     = (unsigned short*)(ws + off_hB);
    float*          cws    = (float*)(ws + off_c);

    // 1) zero feature buffer (padding), h0, c0
    zero_u32_k<<<4096, 256, 0, stream>>>((unsigned int*)xf_bf, (long long)(XF_B / 4));
    zero_u32_k<<<256, 256, 0, stream>>>((unsigned int*)hA, (long long)(H_B / 4));
    zero_u32_k<<<512, 256, 0, stream>>>((unsigned int*)cws, (long long)(C_B / 4));

    // 2) weights -> bf16, combined bias
    cvt_bf16_k<<<4096, 256, 0, stream>>>(wih, wih_bf, (long long)G4 * NH);
    cvt_bf16_k<<<2048, 256, 0, stream>>>(whh, whh_bf, (long long)G4 * HH);
    bias_k<<<(G4 + 255) / 256, 256, 0, stream>>>(bih, bhh, biasws);

    // 3) conv pyramid -> Xf (bf16, left-zero-padded)
    conv_pyramid_k<<<dim3(SS, BB), 256, 0, stream>>>(X, c1w, c1b, c2w, c2b, c3w, c3b, xf_bf);

    // 4) hoisted input projection: Xin = Xf @ W_ih^T + bias  (one big WMMA GEMM)
    gemm_in_k<<<dim3(MROWS / 128, G4 / 128), 256, 0, stream>>>(xf_bf, wih_bf, biasws, xin);

    // 5) 88 recurrent steps (small WMMA GEMM + fused cell)
    unsigned short* hp = hA;
    unsigned short* hn = hB;
    for (int t = 0; t < ML; ++t) {
        lstm_step_k<<<dim3(BB / 64, HH / 16), 128, 0, stream>>>(xin, whh_bf, hp, hn, cws, out, t);
        unsigned short* tmp = hp; hp = hn; hn = tmp;
    }
}
